// SanaBlock_22557168238740
// MI455X (gfx1250) — compile-verified, matrix operand order checked by reading, not run on hardware
//
#include <hip/hip_runtime.h>
#include <hip/hip_bf16.h>

// ---------------------------------------------------------------------------
// SANA transformer block for gfx1250 (MI455X).
// Compute-bound (~610 GFLOP vs ~0.9GB traffic) => all large GEMMs use
// v_wmma_f32_16x16x32_bf16 with fp32 accumulation. A-tile staging uses the
// CDNA5 async global->LDS path (ASYNCcnt). Elementwise/softmax/reduction glue
// runs on VALU.
// ---------------------------------------------------------------------------

typedef unsigned short u16;
typedef __attribute__((ext_vector_type(16))) __bf16 v16bf;
typedef __attribute__((ext_vector_type(8)))  float  v8f;

struct alignas(16) U128 { unsigned int w[4]; };
union  FragU { U128 u[2]; v16bf v; };

#define BB   4
#define NN   4096
#define CC   1152
#define MM   300
#define SAH  36
#define SAD  32
#define CAH  16
#define HD   72
#define IFF  2880
#define ROWS (BB*NN)          // 16384

__device__ __forceinline__ u16 f2bf(float f) {
    unsigned int x = __float_as_uint(f);
    unsigned int r = x + 0x7FFFu + ((x >> 16) & 1u);   // round-to-nearest-even
    return (u16)(r >> 16);
}

// CDNA5 async global->LDS copy, 16B per lane, tracked on ASYNCcnt.
__device__ __forceinline__ void async_ld_b128(unsigned lds_addr, const void* gaddr) {
    asm volatile("global_load_async_to_lds_b128 %0, %1, off"
                 :: "v"(lds_addr), "v"(gaddr) : "memory");
}
__device__ __forceinline__ void wait_async0() {
    asm volatile("s_wait_asynccnt 0x0" ::: "memory");
}

// ---------------------------------------------------------------------------
// fp32 -> bf16 convert
// ---------------------------------------------------------------------------
__global__ void k_cvt(const float* __restrict__ s, u16* __restrict__ d, int n) {
    int i = blockIdx.x * 256 + threadIdx.x;
    if (i < n) d[i] = f2bf(s[i]);
}

// ---------------------------------------------------------------------------
// adaLN table: ada[b, j] = b_ada[j] + sst[j] + sum_k silu(t_emb[b,k]) * w_ada[k,j]
// ---------------------------------------------------------------------------
__global__ void k_ada(const float* __restrict__ t_emb, const float* __restrict__ w_ada,
                      const float* __restrict__ b_ada, const float* __restrict__ sst,
                      float* __restrict__ ada) {
    int i = blockIdx.x * 256 + threadIdx.x;       // 4 * 6912
    if (i >= BB * 6 * CC) return;
    int b = i / (6 * CC);
    int j = i % (6 * CC);
    float s = b_ada[j] + sst[j];
    for (int k = 0; k < CC; ++k) {
        float t = t_emb[b * CC + k];
        float si = t / (1.f + __expf(-t));
        s += si * w_ada[(size_t)k * (6 * CC) + j];
    }
    ada[i] = s;
}

// ---------------------------------------------------------------------------
// LayerNorm + modulation, emits bf16:  out = ln(x) * (one_plus + scale) + shift
// scale/shift indexed [b*bstride + c] (bstride=6*C for adaLN rows, 0 for gamma)
// ---------------------------------------------------------------------------
__global__ __launch_bounds__(256) void k_ln_mod(
    const float* __restrict__ x, u16* __restrict__ out,
    const float* __restrict__ scale, const float* __restrict__ shift,
    int bstride, float one_plus) {
    __shared__ float r1[256], r2[256];
    int row = blockIdx.x;
    int b = row >> 12;                 // row / 4096
    int t = threadIdx.x;
    const float* xr = x + (size_t)row * CC;
    float s = 0.f, s2 = 0.f;
    for (int c = t; c < CC; c += 256) { float v = xr[c]; s += v; s2 += v * v; }
    r1[t] = s; r2[t] = s2;
    __syncthreads();
    for (int st = 128; st > 0; st >>= 1) {
        if (t < st) { r1[t] += r1[t + st]; r2[t] += r2[t + st]; }
        __syncthreads();
    }
    float mean = r1[0] * (1.f / CC);
    float var  = r2[0] * (1.f / CC) - mean * mean;
    float rstd = rsqrtf(var + 1e-6f);
    for (int c = t; c < CC; c += 256) {
        float sc = one_plus + scale[(size_t)b * bstride + c];
        float sh = shift[(size_t)b * bstride + c];
        out[(size_t)row * CC + c] = f2bf((xr[c] - mean) * rstd * sc + sh);
    }
}

// ---------------------------------------------------------------------------
// bf16 WMMA GEMM:  C[M,N] (f32, +bias) = A[M,K] @ B[K,N]   (A,B bf16 bits)
// 128x128 block tile, K-tile 64, 8 waves (2x4), each wave 64x32 via 8 WMMAs.
// A tile staged via global_load_async_to_lds_b128 (ASYNCcnt), B tile staged
// transposed via VGPRs so fragments load as two ds_load_b128 each.
// ---------------------------------------------------------------------------
#define TM 128
#define TN 128
#define TK 64

__global__ __launch_bounds__(256) void k_gemm_bf16(
    const u16* __restrict__ A, const u16* __restrict__ Bw,
    float* __restrict__ Cm, const float* __restrict__ bias,
    int M, int K, int N) {
    __shared__ u16 lsA[TM * TK];        // [m][k]
    __shared__ u16 lsB[TN * TK];        // transposed: [n][k]
    int t    = threadIdx.x;
    int lane = t & 31;
    int w    = t >> 5;
    int wm   = w >> 2;                  // 0..1
    int wn   = w & 3;                   // 0..3
    int bm   = blockIdx.y * TM;
    int bn   = blockIdx.x * TN;

    v8f acc[4][2];
    #pragma unroll
    for (int i = 0; i < 4; ++i)
        #pragma unroll
        for (int j = 0; j < 2; ++j)
            #pragma unroll
            for (int e = 0; e < 8; ++e) acc[i][j][e] = 0.f;

    for (int k0 = 0; k0 < K; k0 += TK) {
        // ---- stage A tile: 128 x 64 bf16, async global->LDS, 16B per lane.
        // Edge rows (gm >= M) are skipped: stale LDS only feeds accumulator
        // rows that the guarded epilogue never stores.
        #pragma unroll
        for (int it = 0; it < 4; ++it) {
            int idx = t + it * 256;
            int row = idx >> 3;
            int kc  = (idx & 7) << 3;
            int gm  = bm + row;
            unsigned lds = (unsigned)(size_t)(void*)&lsA[row * TK + kc];
            if (gm < M)
                async_ld_b128(lds, (const void*)(A + (size_t)gm * K + k0 + kc));
        }
        // ---- stage B tile transposed: 64(k) x 128(n) -> lsB[n][k]
        #pragma unroll
        for (int it = 0; it < 4; ++it) {
            int idx = t + it * 256;
            int kr  = idx >> 4;                 // 0..63
            int nc  = (idx & 15) << 3;          // 0..120
            U128 val = *(const U128*)(Bw + (size_t)(k0 + kr) * N + bn + nc);
            const u16* pv = (const u16*)&val;
            #pragma unroll
            for (int j = 0; j < 8; ++j) lsB[(nc + j) * TK + kr] = pv[j];
        }
        // prefetch next K-stage into caches while we compute
        if (k0 + TK < K) {
            int pr = bm + (t & 127); if (pr >= M) pr = M - 1;
            __builtin_prefetch((const void*)(A + (size_t)pr * K + k0 + TK), 0, 1);
            __builtin_prefetch((const void*)(Bw + (size_t)(k0 + TK + (t >> 2)) * N + bn), 0, 1);
        }
        wait_async0();          // own wave's async LDS writes landed
        __syncthreads();        // cross-wave visibility

        #pragma unroll
        for (int ks = 0; ks < TK; ks += 32) {
            // per-lane K base: lanes 0-15 hold K ks..+7 / ks+16..+23,
            // lanes 16-31 hold K ks+8..+15 / ks+24..+31  (ISA 16-bit A layout)
            int kb = ks + ((lane & 16) >> 1);
            v16bf af[4], bfm[2];
            #pragma unroll
            for (int mt = 0; mt < 4; ++mt) {
                int m = wm * 64 + mt * 16 + (lane & 15);
                FragU f;
                f.u[0] = *(const U128*)&lsA[m * TK + kb];
                f.u[1] = *(const U128*)&lsA[m * TK + kb + 16];
                af[mt] = f.v;
            }
            #pragma unroll
            for (int nt = 0; nt < 2; ++nt) {
                int n = wn * 32 + nt * 16 + (lane & 15);
                FragU f;
                f.u[0] = *(const U128*)&lsB[n * TK + kb];
                f.u[1] = *(const U128*)&lsB[n * TK + kb + 16];
                bfm[nt] = f.v;
            }
            #pragma unroll
            for (int mt = 0; mt < 4; ++mt)
                #pragma unroll
                for (int nt = 0; nt < 2; ++nt)
                    acc[mt][nt] = __builtin_amdgcn_wmma_f32_16x16x32_bf16(
                        false, af[mt], false, bfm[nt], (short)0, acc[mt][nt],
                        false, false);
        }
        __syncthreads();
    }

    // epilogue: C layout — lane = N col, VGPR v -> row v (+8 for lanes 16-31)
    #pragma unroll
    for (int mt = 0; mt < 4; ++mt)
        #pragma unroll
        for (int nt = 0; nt < 2; ++nt) {
            int col   = bn + wn * 32 + nt * 16 + (lane & 15);
            float bia = bias ? bias[col] : 0.f;
            int mbase = bm + wm * 64 + mt * 16 + ((lane & 16) >> 1);
            #pragma unroll
            for (int v = 0; v < 8; ++v) {
                int m = mbase + v;
                if (m < M) Cm[(size_t)m * N + col] = acc[mt][nt][v] + bia;
            }
        }
}

// ---------------------------------------------------------------------------
// Linear attention: kv[b,h,d,e] = sum_n relu(k)[d] * v[e];  ksum[b,h,d]
// qkv layout per row: [3][36][32] (stride 3456)
// ---------------------------------------------------------------------------
__global__ __launch_bounds__(256) void k_la_kv(
    const float* __restrict__ qkv, float* __restrict__ kvb, float* __restrict__ ksb) {
    int bh = blockIdx.x;                 // b*36 + h
    int b  = bh / SAH, h = bh % SAH;
    int t  = threadIdx.x;
    __shared__ float ks[8][SAD], vs[8][SAD];
    int d = t >> 3, e0 = (t & 7) * 4;
    float acc0 = 0.f, acc1 = 0.f, acc2 = 0.f, acc3 = 0.f, ksacc = 0.f;
    const float* base = qkv + (size_t)b * NN * (3 * CC);
    int nn = t >> 5, ln = t & 31;
    for (int n0 = 0; n0 < NN; n0 += 8) {
        size_t r = (size_t)(n0 + nn) * (3 * CC);
        ks[nn][ln] = fmaxf(base[r + CC + h * SAD + ln], 0.f);
        vs[nn][ln] = base[r + 2 * CC + h * SAD + ln];
        __syncthreads();
        #pragma unroll
        for (int q = 0; q < 8; ++q) {
            float kd = ks[q][d];
            acc0 += kd * vs[q][e0 + 0];
            acc1 += kd * vs[q][e0 + 1];
            acc2 += kd * vs[q][e0 + 2];
            acc3 += kd * vs[q][e0 + 3];
        }
        if (t < SAD) {
            #pragma unroll
            for (int q = 0; q < 8; ++q) ksacc += ks[q][t];
        }
        __syncthreads();
    }
    float* kvo = kvb + (size_t)bh * (SAD * SAD);
    kvo[d * SAD + e0 + 0] = acc0;
    kvo[d * SAD + e0 + 1] = acc1;
    kvo[d * SAD + e0 + 2] = acc2;
    kvo[d * SAD + e0 + 3] = acc3;
    if (t < SAD) ksb[(size_t)bh * SAD + t] = ksacc;
}

// out[b,n,h,e] = (relu(q) . kv[:,e]) / (relu(q) . ksum + 1e-6)   -> bf16
__global__ __launch_bounds__(256) void k_la_out(
    const float* __restrict__ qkv, const float* __restrict__ kvb,
    const float* __restrict__ ksb, u16* __restrict__ out) {
    __shared__ float qs[CC];
    __shared__ float den[SAH];
    int row = blockIdx.x, b = row >> 12, t = threadIdx.x;
    const float* qp = qkv + (size_t)row * (3 * CC);
    for (int j = t; j < CC; j += 256) qs[j] = fmaxf(qp[j], 0.f);
    __syncthreads();
    if (t < SAH) {
        const float* kp = ksb + (size_t)(b * SAH + t) * SAD;
        float s = 1e-6f;
        #pragma unroll
        for (int d = 0; d < SAD; ++d) s += qs[t * SAD + d] * kp[d];
        den[t] = s;
    }
    __syncthreads();
    for (int j = t; j < CC; j += 256) {
        int h = j >> 5, e = j & 31;
        const float* kvp = kvb + (size_t)(b * SAH + h) * (SAD * SAD);
        float o = 0.f;
        #pragma unroll
        for (int d = 0; d < SAD; ++d) o += qs[h * SAD + d] * kvp[d * SAD + e];
        out[(size_t)row * CC + j] = f2bf(o / den[h]);
    }
}

// ---------------------------------------------------------------------------
// Cross attention: one thread per (b,h,n). Online softmax over M=300 keys,
// K/V streamed from L2 (kvc layout (B,300,2,16,72)).
// ---------------------------------------------------------------------------
__global__ __launch_bounds__(256) void k_xattn(
    const float* __restrict__ qc, const float* __restrict__ kvc,
    u16* __restrict__ oc) {
    int gid = blockIdx.x * 256 + threadIdx.x;
    if (gid >= BB * CAH * NN) return;
    int n = gid & (NN - 1);
    int h = (gid >> 12) & (CAH - 1);
    int b = gid >> 16;
    const float* q = qc + (size_t)(b * NN + n) * CC + h * HD;
    float qr[HD];
    #pragma unroll
    for (int d = 0; d < HD; ++d) qr[d] = q[d];
    const float scl = rsqrtf((float)HD);
    const float* kbase = kvc + (size_t)b * MM * (2 * CC) + h * HD;
    float mx = -1e30f, ssum = 0.f;
    for (int m = 0; m < MM; ++m) {
        const float* kr = kbase + (size_t)m * (2 * CC);
        float dot = 0.f;
        #pragma unroll
        for (int d = 0; d < HD; ++d) dot += qr[d] * kr[d];
        dot *= scl;
        float nm = fmaxf(mx, dot);
        ssum = ssum * __expf(mx - nm) + __expf(dot - nm);
        mx = nm;
    }
    float o[HD];
    #pragma unroll
    for (int d = 0; d < HD; ++d) o[d] = 0.f;
    const float* vbase = kbase + CC;
    for (int m = 0; m < MM; ++m) {
        const float* kr = kbase + (size_t)m * (2 * CC);
        float dot = 0.f;
        #pragma unroll
        for (int d = 0; d < HD; ++d) dot += qr[d] * kr[d];
        float wgt = __expf(dot * scl - mx);
        const float* vr = vbase + (size_t)m * (2 * CC);
        #pragma unroll
        for (int d = 0; d < HD; ++d) o[d] += wgt * vr[d];
    }
    float inv = 1.f / ssum;
    u16* op = oc + (size_t)(b * NN + n) * CC + h * HD;
    #pragma unroll
    for (int d = 0; d < HD; ++d) op[d] = f2bf(o[d] * inv);
}

// ---------------------------------------------------------------------------
// h = silu(gv[:, :IFF]) * gv[:, IFF:]
// ---------------------------------------------------------------------------
__global__ void k_silu_gate(const float* __restrict__ gv, float* __restrict__ h) {
    size_t i = (size_t)blockIdx.x * 256 + threadIdx.x;
    if (i >= (size_t)ROWS * IFF) return;
    size_t row = i / IFF;
    int    c   = (int)(i % IFF);
    float g = gv[row * (2 * IFF) + c];
    float v = gv[row * (2 * IFF) + IFF + c];
    h[i] = g / (1.f + __expf(-g)) * v;
}

// 3x3 depthwise conv over (B, 64, 64, IFF), SAME padding -> bf16
__global__ void k_dwconv(const float* __restrict__ h, const float* __restrict__ wt,
                         const float* __restrict__ bias, u16* __restrict__ out) {
    size_t i = (size_t)blockIdx.x * 256 + threadIdx.x;
    if (i >= (size_t)ROWS * IFF) return;
    int    c = (int)(i % IFF);
    size_t p = i / IFF;
    int    n = (int)(p & (NN - 1));
    int    b = (int)(p >> 12);
    int y = n >> 6, x = n & 63;
    float s = bias[c];
    #pragma unroll
    for (int ky = 0; ky < 3; ++ky) {
        int yy = y + ky - 1;
        if (yy < 0 || yy > 63) continue;
        #pragma unroll
        for (int kx = 0; kx < 3; ++kx) {
            int xx = x + kx - 1;
            if (xx < 0 || xx > 63) continue;
            s += h[((size_t)b * NN + yy * 64 + xx) * IFF + c] * wt[(ky * 3 + kx) * IFF + c];
        }
    }
    out[i] = f2bf(s);
}

// x_out = x_in + gate[b,c] * y   (gate==nullptr -> 1; y already has bias)
__global__ void k_resid(const float* __restrict__ xin, const float* __restrict__ y,
                        const float* __restrict__ gate, float* __restrict__ xout) {
    size_t i = (size_t)blockIdx.x * 256 + threadIdx.x;
    if (i >= (size_t)ROWS * CC) return;
    int c = (int)(i % CC);
    int b = (int)(i / ((size_t)NN * CC));
    float g = gate ? gate[(size_t)b * (6 * CC) + c] : 1.f;
    xout[i] = xin[i] + g * y[i];
}

// ---------------------------------------------------------------------------
extern "C" void kernel_launch(void* const* d_in, const int* in_sizes, int n_in,
                              void* d_out, int out_size, void* d_ws, size_t ws_size,
                              hipStream_t stream) {
    (void)in_sizes; (void)n_in; (void)out_size; (void)ws_size;
    const float* x        = (const float*)d_in[0];
    const float* cond     = (const float*)d_in[1];
    const float* t_emb    = (const float*)d_in[2];
    const float* w_qkv    = (const float*)d_in[3];
    const float* w_sa_out = (const float*)d_in[4];
    const float* b_sa_out = (const float*)d_in[5];
    const float* gamma_c  = (const float*)d_in[6];
    const float* beta_c   = (const float*)d_in[7];
    const float* w_q      = (const float*)d_in[8];
    const float* b_q      = (const float*)d_in[9];
    const float* w_kv     = (const float*)d_in[10];
    const float* b_kv     = (const float*)d_in[11];
    const float* w_co     = (const float*)d_in[12];
    const float* b_co     = (const float*)d_in[13];
    const float* w_fc1    = (const float*)d_in[14];
    const float* b_fc1    = (const float*)d_in[15];
    const float* dw_k     = (const float*)d_in[16];
    const float* dw_b     = (const float*)d_in[17];
    const float* w_fc2    = (const float*)d_in[18];
    const float* b_fc2    = (const float*)d_in[19];
    const float* w_ada    = (const float*)d_in[20];
    const float* b_ada    = (const float*)d_in[21];
    const float* sst      = (const float*)d_in[22];
    float* out = (float*)d_out;

    // ---- carve workspace
    char* ws = (char*)d_ws;
    size_t off = 0;
    auto carve = [&](size_t bytes) -> void* {
        off = (off + 255) & ~(size_t)255;
        void* p = ws + off;
        off += bytes;
        return p;
    };
    u16* wb_qkv = (u16*)carve((size_t)CC * 3 * CC * 2);
    u16* wb_sa  = (u16*)carve((size_t)CC * CC * 2);
    u16* wb_q   = (u16*)carve((size_t)CC * CC * 2);
    u16* wb_kv  = (u16*)carve((size_t)CC * 2 * CC * 2);
    u16* wb_co  = (u16*)carve((size_t)CC * CC * 2);
    u16* wb_fc1 = (u16*)carve((size_t)CC * 2 * IFF * 2);
    u16* wb_fc2 = (u16*)carve((size_t)IFF * CC * 2);
    u16* cond_bf= (u16*)carve((size_t)BB * MM * CC * 2);
    float* ada  = (float*)carve((size_t)BB * 6 * CC * 4);
    u16* act_bf = (u16*)carve((size_t)ROWS * IFF * 2);            // bf16 activations (reused)
    float* G    = (float*)carve((size_t)ROWS * 2 * IFF * 4);      // big fp32 GEMM output (reused)
    float* kvc  = (float*)carve((size_t)BB * MM * 2 * CC * 4);    // cross K/V
    float* xcur = (float*)carve((size_t)ROWS * CC * 4);           // running residual
    float* hbuf = (float*)carve((size_t)ROWS * IFF * 4);          // FF hidden
    float* lakv = (float*)carve((size_t)BB * SAH * SAD * SAD * 4);
    float* laks = (float*)carve((size_t)BB * SAH * SAD * 4);

    auto cvt = [&](const float* s, u16* d, size_t n) {
        k_cvt<<<dim3((unsigned)((n + 255) / 256)), 256, 0, stream>>>(s, d, (int)n);
    };
    auto gemm = [&](const u16* A, const u16* Bw, float* Cm, const float* bias,
                    int M, int K, int N) {
        dim3 grid((unsigned)(N / TN), (unsigned)((M + TM - 1) / TM));
        k_gemm_bf16<<<grid, 256, 0, stream>>>(A, Bw, Cm, bias, M, K, N);
    };
    const unsigned GRID_RC  = (unsigned)(((size_t)ROWS * CC + 255) / 256);
    const unsigned GRID_RF  = (unsigned)(((size_t)ROWS * IFF + 255) / 256);

    // ---- weight / conditioning conversion to bf16
    cvt(w_qkv,    wb_qkv, (size_t)CC * 3 * CC);
    cvt(w_sa_out, wb_sa,  (size_t)CC * CC);
    cvt(w_q,      wb_q,   (size_t)CC * CC);
    cvt(w_kv,     wb_kv,  (size_t)CC * 2 * CC);
    cvt(w_co,     wb_co,  (size_t)CC * CC);
    cvt(w_fc1,    wb_fc1, (size_t)CC * 2 * IFF);
    cvt(w_fc2,    wb_fc2, (size_t)IFF * CC);
    cvt(cond,     cond_bf,(size_t)BB * MM * CC);

    // ---- adaLN
    k_ada<<<dim3((BB * 6 * CC + 255) / 256), 256, 0, stream>>>(t_emb, w_ada, b_ada, sst, ada);

    // ---- self-attention branch
    k_ln_mod<<<ROWS, 256, 0, stream>>>(x, act_bf, ada + CC, ada, 6 * CC, 1.f); // shift0, scale1
    gemm(act_bf, wb_qkv, G, nullptr, ROWS, CC, 3 * CC);                        // qkv
    k_la_kv<<<BB * SAH, 256, 0, stream>>>(G, lakv, laks);
    k_la_out<<<ROWS, 256, 0, stream>>>(G, lakv, laks, act_bf);
    gemm(act_bf, wb_sa, G, b_sa_out, ROWS, CC, CC);
    k_resid<<<GRID_RC, 256, 0, stream>>>(x, G, ada + 2 * CC, xcur);            // gate_sa

    // ---- cross-attention branch
    k_ln_mod<<<ROWS, 256, 0, stream>>>(xcur, act_bf, gamma_c, beta_c, 0, 0.f);
    gemm(act_bf, wb_q, G, b_q, ROWS, CC, CC);                                  // qc
    gemm(cond_bf, wb_kv, kvc, b_kv, BB * MM, CC, 2 * CC);                      // kc,vc
    k_xattn<<<(BB * CAH * NN) / 256, 256, 0, stream>>>(G, kvc, act_bf);        // oc -> bf16
    gemm(act_bf, wb_co, G, b_co, ROWS, CC, CC);
    k_resid<<<GRID_RC, 256, 0, stream>>>(xcur, G, nullptr, xcur);

    // ---- feed-forward branch
    k_ln_mod<<<ROWS, 256, 0, stream>>>(xcur, act_bf, ada + 4 * CC, ada + 3 * CC, 6 * CC, 1.f);
    gemm(act_bf, wb_fc1, G, b_fc1, ROWS, CC, 2 * IFF);
    k_silu_gate<<<GRID_RF, 256, 0, stream>>>(G, hbuf);
    k_dwconv<<<GRID_RF, 256, 0, stream>>>(hbuf, dw_k, dw_b, act_bf);
    gemm(act_bf, wb_fc2, G, b_fc2, ROWS, IFF, CC);
    k_resid<<<GRID_RC, 256, 0, stream>>>(xcur, G, ada + 5 * CC, out);          // gate_ff
}
